// GAT_71983651881465
// MI455X (gfx1250) — matrix operand order repaired and checked by reading, not compiled, not run on hardware
//
#include <hip/hip_runtime.h>
#include <hip/hip_bf16.h>

// ---------------------------------------------------------------------------
// GAT forward for MI455X (gfx1250, wave32).
// Dense x@W via V_WMMA_F32_16X16X32_F16, 2x4 tile blocking per wave
// (8 accumulators = 64 VGPRs, no spills), f16-packed A (BN fused into the
// layer-1 pack) and fragment-packed B.
// Edge softmax/aggregation via L2-resident float atomics.
// ---------------------------------------------------------------------------

typedef __attribute__((ext_vector_type(16))) _Float16 v16h;
typedef __attribute__((ext_vector_type(8)))  _Float16 v8h;
typedef __attribute__((ext_vector_type(8)))  float    v8f;

#define ATT_SLOPE 0.2f
#define ACT_SLOPE 0.01f
#define BN_EPS 1e-5f

static __device__ __forceinline__ void atomicMaxF(float* addr, float v) {
    // Sign-split trick: non-negative -> int max; negative -> unsigned min.
    // Correct for mixed signs with init = -1e30f.
    if (v >= 0.0f) atomicMax((int*)addr, __float_as_int(v));
    else           atomicMin((unsigned int*)addr, (unsigned int)__float_as_int(v));
}

// ---------------------------- elementwise helpers ---------------------------

__global__ void k_fill(float* __restrict__ p, size_t n, float v) {
    size_t i = (size_t)blockIdx.x * blockDim.x + threadIdx.x;
    if (i < n) p[i] = v;
}

// BatchNorm (F=128) fused with f32->f16 conversion for the GEMM-1 A operand.
__global__ void k_bn_f16(const float* __restrict__ x,
                         const float* __restrict__ g, const float* __restrict__ b,
                         const float* __restrict__ mu, const float* __restrict__ var,
                         _Float16* __restrict__ out, size_t n) {
    size_t i = (size_t)blockIdx.x * blockDim.x + threadIdx.x;
    if (i >= n) return;
    int c = (int)(i & 127);
    float v = (x[i] - mu[c]) * rsqrtf(var[c] + BN_EPS) * g[c] + b[c];
    out[i] = (_Float16)v;
}

__global__ void k_cvt_f16(const float* __restrict__ in, _Float16* __restrict__ out,
                          size_t n) {
    size_t i = (size_t)blockIdx.x * blockDim.x + threadIdx.x;
    if (i < n) out[i] = (_Float16)in[i];
}

__global__ void k_bias_act(float* __restrict__ p, const float* __restrict__ bias,
                           size_t n) {
    size_t i = (size_t)blockIdx.x * blockDim.x + threadIdx.x;
    if (i >= n) return;
    int j = (int)(i & 511);  // HC = 512
    float v = p[i] + bias[j];
    p[i] = v > 0.0f ? v : ACT_SLOPE * v;
}

// ------------------------------- WMMA GEMM ---------------------------------

// Pack W [K x Nout] (f32, row-major) into per-wave B-fragment layout (f16):
// index ((kt*nTilesN + tn)*32 + lane)*16 + j holds
// W[kt*32 + (lane<16?0:8) + (j<8?j:j+8)][tn*16 + lane%16].
__global__ void k_pack_w(const float* __restrict__ W, _Float16* __restrict__ Bp,
                         int K, int Nout) {
    int tid = blockIdx.x * blockDim.x + threadIdx.x;
    int nTilesN = Nout >> 4;
    int total = (K >> 5) * nTilesN * 32;
    if (tid >= total) return;
    int lane = tid & 31;
    int t    = tid >> 5;
    int tn   = t % nTilesN;
    int kt   = t / nTilesN;
    int n    = tn * 16 + (lane & 15);
    int kb   = kt * 32 + ((lane < 16) ? 0 : 8);
    _Float16* o = Bp + (size_t)tid * 16;
#pragma unroll
    for (int j = 0; j < 16; ++j)
        o[j] = (_Float16)W[(size_t)(kb + (j < 8 ? j : j + 8)) * Nout + n];
}

// C[Nrows x Nout] = A[Nrows x K] @ W (packed). A is row-major f16.
// One 32x64 output block per wave: 2x4 tiles of 16x16, 8 WMMAs per K-step.
// 8 accs (64 VGPRs) + 2 A frags (16) + live B frag (8) ~= 110 VGPRs: no spill.
__global__ void __launch_bounds__(256, 1)
k_gemm_wmma(const _Float16* __restrict__ A,
            const _Float16* __restrict__ Bp,
            float* __restrict__ C,
            int Nrows, int K, int Nout) {
    int lane = threadIdx.x & 31;
    int wid  = (blockIdx.x * blockDim.x + threadIdx.x) >> 5;
    int nTilesN = Nout >> 4;
    int nGrpM = (((Nrows + 15) >> 4) + 1) >> 1;   // M groups of 2 tiles (ragged ok)
    int nGrpN = nTilesN >> 2;                      // N groups of 4 tiles (Nout%64==0)
    if (wid >= nGrpM * nGrpN) return;              // wave-uniform exit
    int gm = wid / nGrpN;
    int gn = wid % nGrpN;

    int lr   = lane & 15;
    int koff = (lane < 16) ? 0 : 8;                // ISA A-fragment K base
    int mrow[2];
#pragma unroll
    for (int i = 0; i < 2; ++i) {
        int m = gm * 32 + i * 16 + lr;
        mrow[i] = (m < Nrows) ? m : (Nrows - 1);   // clamp keeps EXEC all-1s
    }

    v8f acc[2][4];
#pragma unroll
    for (int i = 0; i < 2; ++i)
#pragma unroll
        for (int j = 0; j < 4; ++j) acc[i][j] = (v8f){};

    int nkt = K >> 5;
#pragma unroll 1
    for (int kt = 0; kt < nkt; ++kt) {
        v16h a[2];
#pragma unroll
        for (int i = 0; i < 2; ++i) {
            // halves 0..7 = K kt*32+koff .. +7 ; halves 8..15 = +16 .. +23
            const _Float16* ap = A + (size_t)mrow[i] * K + (size_t)kt * 32 + koff;
            v8h lo = *(const v8h*)ap;
            v8h hi = *(const v8h*)(ap + 16);
#pragma unroll
            for (int q = 0; q < 8; ++q) { a[i][q] = lo[q]; a[i][8 + q] = hi[q]; }
        }
#pragma unroll
        for (int j = 0; j < 4; ++j) {
            int tn = gn * 4 + j;
            const _Float16* bp = Bp + ((size_t)(kt * nTilesN + tn) * 32 + lane) * 16;
            v8h b0 = *(const v8h*)bp;
            v8h b1 = *(const v8h*)(bp + 8);
            v16h bfrag;
#pragma unroll
            for (int q = 0; q < 8; ++q) { bfrag[q] = b0[q]; bfrag[8 + q] = b1[q]; }
#pragma unroll
            for (int i = 0; i < 2; ++i)
                acc[i][j] = __builtin_amdgcn_wmma_f32_16x16x32_f16(
                    false, a[i], false, bfrag, (short)0, acc[i][j], false, false);
        }
    }

    // C/D layout: VGPR r -> row r (lanes 0-15) or row r+8 (lanes 16-31)
    int rhalf = (lane < 16) ? 0 : 8;
#pragma unroll
    for (int i = 0; i < 2; ++i) {
        int rbase = gm * 32 + i * 16 + rhalf;
#pragma unroll
        for (int j = 0; j < 4; ++j) {
            int col = gn * 64 + j * 16 + lr;
#pragma unroll
            for (int r = 0; r < 8; ++r) {
                int row = rbase + r;
                if (row < Nrows) C[(size_t)row * Nout + col] = acc[i][j][r];
            }
        }
    }
}

// ------------------------- attention (H=4, C=128) --------------------------

__global__ void k_att(const float* __restrict__ Hbuf,
                      const float* __restrict__ attS, const float* __restrict__ attD,
                      float* __restrict__ aS, float* __restrict__ aD, int N) {
    int tid = blockIdx.x * blockDim.x + threadIdx.x;
    if (tid >= N * 4) return;
    int n = tid >> 2, h = tid & 3;
    const float* hp = Hbuf + (size_t)n * 512 + h * 128;
    const float* sp = attS + h * 128;
    const float* dp = attD + h * 128;
    float s = 0.0f, d = 0.0f;
#pragma unroll 4
    for (int c = 0; c < 128; ++c) { float v = hp[c]; s += v * sp[c]; d += v * dp[c]; }
    aS[tid] = s;
    aD[tid] = d;
}

// Edge passes (per (edge, head)); edges [0,E) real, [E, E+N) self-loops.
__global__ void k_edge_max(const int* __restrict__ esrc, const int* __restrict__ edst,
                           int E, int N,
                           const float* __restrict__ aS, const float* __restrict__ aD,
                           float* __restrict__ emax, float* __restrict__ ebuf) {
    int tid = blockIdx.x * blockDim.x + threadIdx.x;
    int Etot = E + N;
    if (tid >= Etot * 4) return;
    int e = tid >> 2, h = tid & 3;
    int s = (e < E) ? esrc[e] : (e - E);
    int d = (e < E) ? edst[e] : (e - E);
    float v = aS[s * 4 + h] + aD[d * 4 + h];
    v = v > 0.0f ? v : ATT_SLOPE * v;
    ebuf[tid] = v;
    atomicMaxF(&emax[d * 4 + h], v);
}

__global__ void k_edge_sum(const int* __restrict__ esrc, const int* __restrict__ edst,
                           int E, int N,
                           const float* __restrict__ emax,
                           float* __restrict__ ebuf, float* __restrict__ den) {
    int tid = blockIdx.x * blockDim.x + threadIdx.x;
    int Etot = E + N;
    if (tid >= Etot * 4) return;
    int e = tid >> 2, h = tid & 3;
    int d = (e < E) ? edst[e] : (e - E);
    float w = __expf(ebuf[tid] - emax[d * 4 + h]);
    ebuf[tid] = w;
    atomicAdd(&den[d * 4 + h], w);
    (void)esrc;
}

// One wave per edge; lanes stride the 512 channels.
__global__ void k_edge_aggr(const int* __restrict__ esrc, const int* __restrict__ edst,
                            int E, int N,
                            const float* __restrict__ ebuf, const float* __restrict__ den,
                            const float* __restrict__ Hbuf, float* __restrict__ out) {
    int wid  = (blockIdx.x * blockDim.x + threadIdx.x) >> 5;
    int lane = threadIdx.x & 31;
    int Etot = E + N;
    if (wid >= Etot) return;
    int s = (wid < E) ? esrc[wid] : (wid - E);
    int d = (wid < E) ? edst[wid] : (wid - E);
    float alpha[4];
#pragma unroll
    for (int h = 0; h < 4; ++h)
        alpha[h] = ebuf[wid * 4 + h] / (den[d * 4 + h] + 1e-16f);
    const float* src = Hbuf + (size_t)s * 512;
    float* dst = out + (size_t)d * 512;
#pragma unroll
    for (int c = lane; c < 512; c += 32)
        atomicAdd(&dst[c], src[c] * alpha[c >> 7]);
}

// ------------------------------ layer 3 (H=1,C=2) --------------------------

__global__ void k_l3_gemm_att(const float* __restrict__ A, const float* __restrict__ W3,
                              const float* __restrict__ as, const float* __restrict__ ad,
                              float* __restrict__ h3, float* __restrict__ aS3,
                              float* __restrict__ aD3, int N) {
    int n = blockIdx.x * blockDim.x + threadIdx.x;
    if (n >= N) return;
    const float* a = A + (size_t)n * 512;
    float h0 = 0.0f, h1 = 0.0f;
#pragma unroll 4
    for (int k = 0; k < 512; ++k) {
        float av = a[k];
        h0 += av * W3[k * 2 + 0];
        h1 += av * W3[k * 2 + 1];
    }
    h3[n * 2 + 0] = h0;
    h3[n * 2 + 1] = h1;
    aS3[n] = h0 * as[0] + h1 * as[1];
    aD3[n] = h0 * ad[0] + h1 * ad[1];
}

__global__ void k_edge3_max(const int* __restrict__ esrc, const int* __restrict__ edst,
                            int E, int N, const float* __restrict__ aS,
                            const float* __restrict__ aD,
                            float* __restrict__ emax, float* __restrict__ ebuf) {
    int e = blockIdx.x * blockDim.x + threadIdx.x;
    if (e >= E + N) return;
    int s = (e < E) ? esrc[e] : (e - E);
    int d = (e < E) ? edst[e] : (e - E);
    float v = aS[s] + aD[d];
    v = v > 0.0f ? v : ATT_SLOPE * v;
    ebuf[e] = v;
    atomicMaxF(&emax[d], v);
}

__global__ void k_edge3_sum(const int* __restrict__ esrc, const int* __restrict__ edst,
                            int E, int N, const float* __restrict__ emax,
                            float* __restrict__ ebuf, float* __restrict__ den) {
    int e = blockIdx.x * blockDim.x + threadIdx.x;
    if (e >= E + N) return;
    int d = (e < E) ? edst[e] : (e - E);
    float w = __expf(ebuf[e] - emax[d]);
    ebuf[e] = w;
    atomicAdd(&den[d], w);
    (void)esrc;
}

__global__ void k_edge3_aggr(const int* __restrict__ esrc, const int* __restrict__ edst,
                             int E, int N, const float* __restrict__ ebuf,
                             const float* __restrict__ den,
                             const float* __restrict__ h3, float* __restrict__ acc) {
    int e = blockIdx.x * blockDim.x + threadIdx.x;
    if (e >= E + N) return;
    int s = (e < E) ? esrc[e] : (e - E);
    int d = (e < E) ? edst[e] : (e - E);
    float alpha = ebuf[e] / (den[d] + 1e-16f);
    atomicAdd(&acc[d * 2 + 0], h3[s * 2 + 0] * alpha);
    atomicAdd(&acc[d * 2 + 1], h3[s * 2 + 1] * alpha);
}

__global__ void k_final(const float* __restrict__ acc, const float* __restrict__ b3,
                        const float* __restrict__ temp, float* __restrict__ out,
                        int n) {
    int i = blockIdx.x * blockDim.x + threadIdx.x;
    if (i >= n) return;
    float t = temp[0];
    t = fminf(fmaxf(t, 0.1f), 2.0f);
    out[i] = (acc[i] + b3[i & 1]) / (t + 1e-8f);
}

// ------------------------------- host driver -------------------------------

static inline int cdiv(long long a, long long b) { return (int)((a + b - 1) / b); }

extern "C" void kernel_launch(void* const* d_in, const int* in_sizes, int n_in,
                              void* d_out, int out_size, void* d_ws, size_t ws_size,
                              hipStream_t stream) {
    const float* x       = (const float*)d_in[0];
    const int*   ei      = (const int*)d_in[1];
    const float* bn_g    = (const float*)d_in[2];
    const float* bn_b    = (const float*)d_in[3];
    const float* bn_m    = (const float*)d_in[4];
    const float* bn_v    = (const float*)d_in[5];
    const float* W1      = (const float*)d_in[6];
    const float* as1     = (const float*)d_in[7];
    const float* ad1     = (const float*)d_in[8];
    const float* b1      = (const float*)d_in[9];
    const float* W2      = (const float*)d_in[10];
    const float* as2     = (const float*)d_in[11];
    const float* ad2     = (const float*)d_in[12];
    const float* b2      = (const float*)d_in[13];
    const float* W3      = (const float*)d_in[14];
    const float* as3v    = (const float*)d_in[15];
    const float* ad3v    = (const float*)d_in[16];
    const float* b3      = (const float*)d_in[17];
    const float* temp    = (const float*)d_in[18];

    const int N = in_sizes[0] / 128;
    const int E = in_sizes[1] / 2;
    const int Etot = E + N;
    const int* esrc = ei;
    const int* edst = ei + E;

    // workspace carve-up (float units, 64B-padded)
    float* ws = (float*)d_ws;
    size_t o = 0;
    auto alloc = [&](size_t cnt) { float* p = ws + o; o += (cnt + 15) & ~(size_t)15; return p; };
    _Float16* Ah = (_Float16*)alloc((size_t)N * 512 / 2);   // f16 A operand (N*512 halves)
    float* bufH  = alloc((size_t)N * 512);
    float* bufO  = alloc((size_t)N * 512);
    float* aS    = alloc((size_t)N * 4);
    float* aD    = alloc((size_t)N * 4);
    float* emax  = alloc((size_t)N * 4);
    float* den   = alloc((size_t)N * 4);
    float* ebuf  = alloc((size_t)Etot * 4);
    _Float16* packW = (_Float16*)alloc((size_t)512 * 512 / 2);  // 512*512 halves
    float* h3    = alloc((size_t)N * 2);
    float* aS3   = alloc((size_t)N);
    float* aD3   = alloc((size_t)N);
    float* emax3 = alloc((size_t)N);
    float* den3  = alloc((size_t)N);
    float* acc3  = alloc((size_t)N * 2);
    (void)ws_size; (void)n_in; (void)out_size;

    const int B = 256;
    auto gemm_waves = [&](int Nout) {
        int nGrpM = (((N + 15) >> 4) + 1) >> 1;
        int nGrpN = (Nout >> 4) >> 2;
        return (long long)nGrpM * nGrpN * 32;
    };

    // ---- layer 1: K=128, HC=512 ----
    {
        // BN fused into f16 A-pack
        k_bn_f16<<<cdiv((long long)N * 128, B), B, 0, stream>>>(x, bn_g, bn_b, bn_m, bn_v,
                                                               Ah, (size_t)N * 128);
        int packThreads = (128 / 32) * (512 / 16) * 32;
        k_pack_w<<<cdiv(packThreads, B), B, 0, stream>>>(W1, packW, 128, 512);
        k_gemm_wmma<<<cdiv(gemm_waves(512), B), B, 0, stream>>>(Ah, packW, bufH, N, 128, 512);
        k_att<<<cdiv((long long)N * 4, B), B, 0, stream>>>(bufH, as1, ad1, aS, aD, N);
        k_fill<<<cdiv((long long)N * 4, B), B, 0, stream>>>(emax, (size_t)N * 4, -1e30f);
        k_fill<<<cdiv((long long)N * 4, B), B, 0, stream>>>(den, (size_t)N * 4, 0.0f);
        k_fill<<<cdiv((long long)N * 512, B), B, 0, stream>>>(bufO, (size_t)N * 512, 0.0f);
        k_edge_max<<<cdiv((long long)Etot * 4, B), B, 0, stream>>>(esrc, edst, E, N, aS, aD, emax, ebuf);
        k_edge_sum<<<cdiv((long long)Etot * 4, B), B, 0, stream>>>(esrc, edst, E, N, emax, ebuf, den);
        k_edge_aggr<<<cdiv((long long)Etot * 32, B), B, 0, stream>>>(esrc, edst, E, N, ebuf, den, bufH, bufO);
        k_bias_act<<<cdiv((long long)N * 512, B), B, 0, stream>>>(bufO, b1, (size_t)N * 512);
    }

    // ---- layer 2: K=512, HC=512 ----
    {
        k_cvt_f16<<<cdiv((long long)N * 512, B), B, 0, stream>>>(bufO, Ah, (size_t)N * 512);
        int packThreads = (512 / 32) * (512 / 16) * 32;
        k_pack_w<<<cdiv(packThreads, B), B, 0, stream>>>(W2, packW, 512, 512);
        k_gemm_wmma<<<cdiv(gemm_waves(512), B), B, 0, stream>>>(Ah, packW, bufH, N, 512, 512);
        // bufO (layer-1 output) is consumed; reuse as layer-2 aggregation target
        k_fill<<<cdiv((long long)N * 512, B), B, 0, stream>>>(bufO, (size_t)N * 512, 0.0f);
        k_att<<<cdiv((long long)N * 4, B), B, 0, stream>>>(bufH, as2, ad2, aS, aD, N);
        k_fill<<<cdiv((long long)N * 4, B), B, 0, stream>>>(emax, (size_t)N * 4, -1e30f);
        k_fill<<<cdiv((long long)N * 4, B), B, 0, stream>>>(den, (size_t)N * 4, 0.0f);
        k_edge_max<<<cdiv((long long)Etot * 4, B), B, 0, stream>>>(esrc, edst, E, N, aS, aD, emax, ebuf);
        k_edge_sum<<<cdiv((long long)Etot * 4, B), B, 0, stream>>>(esrc, edst, E, N, emax, ebuf, den);
        k_edge_aggr<<<cdiv((long long)Etot * 32, B), B, 0, stream>>>(esrc, edst, E, N, ebuf, den, bufH, bufO);
        k_bias_act<<<cdiv((long long)N * 512, B), B, 0, stream>>>(bufO, b2, (size_t)N * 512);
    }

    // ---- layer 3: K=512 -> 2, H=1 ----
    {
        k_l3_gemm_att<<<cdiv(N, B), B, 0, stream>>>(bufO, W3, as3v, ad3v, h3, aS3, aD3, N);
        k_fill<<<cdiv(N, B), B, 0, stream>>>(emax3, (size_t)N, -1e30f);
        k_fill<<<cdiv(N, B), B, 0, stream>>>(den3, (size_t)N, 0.0f);
        k_fill<<<cdiv((long long)N * 2, B), B, 0, stream>>>(acc3, (size_t)N * 2, 0.0f);
        k_edge3_max<<<cdiv(Etot, B), B, 0, stream>>>(esrc, edst, E, N, aS3, aD3, emax3, ebuf);
        k_edge3_sum<<<cdiv(Etot, B), B, 0, stream>>>(esrc, edst, E, N, emax3, ebuf, den3);
        k_edge3_aggr<<<cdiv(Etot, B), B, 0, stream>>>(esrc, edst, E, N, ebuf, den3, h3, acc3);
        k_final<<<cdiv((long long)N * 2, B), B, 0, stream>>>(acc3, b3, temp, (float*)d_out, N * 2);
    }
}
